// RelPositionMultiHeadedAttention_62070867362193
// MI455X (gfx1250) — matrix operand order repaired and checked by reading, not compiled, not run on hardware
//
#include <hip/hip_runtime.h>

typedef __attribute__((ext_vector_type(16))) __bf16 v16bf;
typedef __attribute__((ext_vector_type(8)))  float  v8f;
typedef __attribute__((ext_vector_type(4)))  unsigned int v4u;
typedef __attribute__((ext_vector_type(8)))  int v8i;
typedef __attribute__((ext_vector_type(4)))  int v4i;
typedef unsigned short u16;
typedef unsigned int   u32;
typedef unsigned long long u64;

#define N_HEADC 8
#define N_FEATC 512
#define D_KC    64
#define BATCHC  4
#define SEQC    2048
#define BTC     (BATCHC * SEQC)

enum { MODE_QU = 0, MODE_KT = 1, MODE_V = 2, MODE_PT = 3, MODE_OUT = 4 };

__device__ __forceinline__ u16 f2bf(float f) {
  u32 u = __float_as_uint(f);
  u32 r = u + 0x7FFFu + ((u >> 16) & 1u);   // round-to-nearest-even
  return (u16)(r >> 16);
}

union FragBF { uint4 q[2]; v16bf v; };

// A-operand (16x32 bf16): lane&15 = M row; K interleave per ISA:
// lane<16 -> K[0..7] at +0B, K[16..23] at +32B ; lane>=16 -> +16B / +48B
__device__ __forceinline__ v16bf load_frag_a(const char* rowbase, int lane) {
  FragBF f;
  const uint4* p = (const uint4*)(rowbase + ((lane >> 4) << 4));
  f.q[0] = p[0];
  f.q[1] = p[2];          // +32 bytes
  return f.v;
}

// B-operand (32x16 bf16): lane&15 = N col; K per ISA table order, row stride ld elems
__device__ __forceinline__ v16bf load_frag_b(const char* base, int lane, int ld) {
  union { u16 u[16]; v16bf v; } f;
  const u16* p = (const u16*)base + (lane & 15);
  int hk = (lane >> 4) << 3;
#pragma unroll
  for (int s = 0; s < 16; ++s) {
    int k = (s < 8) ? (hk + s) : (16 + hk + s - 8);
    f.u[s] = p[(size_t)k * ld];
  }
  return f.v;
}

__device__ __forceinline__ v8f wmma_bf16(v16bf a, v16bf b, v8f c) {
  return __builtin_amdgcn_wmma_f32_16x16x32_bf16(false, a, false, b, (short)0, c,
                                                 false, false);
}

// TDM: 2D bf16 tile load, 64x64 tile, LDS padding 16B per 128B row -> 144B stride
__device__ __forceinline__ void tdm_load_2d(u32 lds_addr, const void* gptr,
                                            u32 row_stride_elems) {
  u64 ga = (u64)(uintptr_t)gptr;
  v4u g0;
  g0[0] = 1u;                                               // count = 1 (valid)
  g0[1] = lds_addr;                                         // LDS byte address
  g0[2] = (u32)ga;                                          // global addr lo
  g0[3] = ((u32)(ga >> 32) & 0x01FFFFFFu) | 0x80000000u;    // addr hi | type=2
  v8i g1;
  // data_size=2B (1<<16) | pad_enable (1<<20) | pad_interval=128B (4<<22)
  // | pad_amount=16B (3<<25)
  g1[0] = (int)0x07110000u;
  g1[1] = (int)(64u << 16);        // tensor_dim0 = 64 (lo16 in [31:16])
  g1[2] = (int)(64u << 16);        // tensor_dim1 = 64
  g1[3] = (int)(64u << 16);        // tile_dim0 = 64
  g1[4] = 64;                      // tile_dim1 = 64, tile_dim2 = 0
  g1[5] = (int)row_stride_elems;   // tensor_dim0_stride (lo32)
  g1[6] = 0;
  g1[7] = 0;
  v4i gz4 = {0, 0, 0, 0};
  v8i gz8 = {0, 0, 0, 0, 0, 0, 0, 0};
  __builtin_amdgcn_tensor_load_to_lds(g0, g1, gz4, gz4, gz8, 0);
}

// ---------------- fp32 -> bf16 converters ----------------
__global__ void cvt_bf16(const float* __restrict__ src, u16* __restrict__ dst, int n) {
  int i = blockIdx.x * blockDim.x + threadIdx.x;
  int stride = gridDim.x * blockDim.x;
  for (; i < n; i += stride) dst[i] = f2bf(src[i]);
}

// W (N x K) fp32 row-major  ->  Wt (K x N) bf16  (for WMMA B-operand)
__global__ void cvt_bf16_tr(const float* __restrict__ src, u16* __restrict__ dst) {
  int i = blockIdx.x * blockDim.x + threadIdx.x;
  if (i >= N_FEATC * N_FEATC) return;
  int n = i >> 9;        // row of W
  int k = i & 511;       // col of W
  dst[(size_t)k * N_FEATC + n] = f2bf(src[i]);
}

// ---------------- bf16 WMMA GEMM:  C[M x 512] = A[M x 512] * Wt + bias ----------------
__global__ __launch_bounds__(256) void gemm_bf16_tn(
    const u16* __restrict__ A, const u16* __restrict__ Wt,
    const float* __restrict__ bias, int mode,
    u16* __restrict__ out0, u16* __restrict__ out1,
    const float* __restrict__ pbu, const float* __restrict__ pbv,
    float* __restrict__ outf)
{
  const int lane = threadIdx.x & 31;
  const int wave = threadIdx.x >> 5;
  const int wm = wave >> 2;
  const int wn = wave & 3;
  const int m0 = blockIdx.x * 128 + wm * 64;
  const int n0 = blockIdx.y * 128 + wn * 32;

  v8f acc[4][2];
#pragma unroll
  for (int mi = 0; mi < 4; ++mi)
#pragma unroll
    for (int ni = 0; ni < 2; ++ni)
      acc[mi][ni] = (v8f){0.f, 0.f, 0.f, 0.f, 0.f, 0.f, 0.f, 0.f};

  const char* Ab = (const char*)A;
  const char* Bb = (const char*)Wt;

#pragma unroll 2
  for (int kq = 0; kq < 16; ++kq) {
    v16bf af[4];
#pragma unroll
    for (int mi = 0; mi < 4; ++mi) {
      size_t row = (size_t)(m0 + mi * 16 + (lane & 15));
      const char* p = Ab + row * 1024 + (size_t)kq * 64;
      if (mi == 0 && kq < 15) __builtin_prefetch(p + 64, 0, 1);
      af[mi] = load_frag_a(p, lane);
    }
    v16bf bf[2];
#pragma unroll
    for (int ni = 0; ni < 2; ++ni) {
      size_t krow = (size_t)(kq * 32);
      bf[ni] = load_frag_b(Bb + (krow * 512 + (size_t)(n0 + ni * 16)) * 2, lane, 512);
    }
#pragma unroll
    for (int mi = 0; mi < 4; ++mi)
#pragma unroll
      for (int ni = 0; ni < 2; ++ni)
        acc[mi][ni] = wmma_bf16(af[mi], bf[ni], acc[mi][ni]);
  }

#pragma unroll
  for (int mi = 0; mi < 4; ++mi) {
#pragma unroll
    for (int ni = 0; ni < 2; ++ni) {
#pragma unroll
      for (int r = 0; r < 8; ++r) {
        int m = m0 + mi * 16 + ((lane >> 4) << 3) + r;
        int o = n0 + ni * 16 + (lane & 15);
        float val = acc[mi][ni][r];
        if (bias) val += bias[o];
        if (mode == MODE_QU) {
          int b = m >> 11, t = m & 2047;
          int h = o >> 6, d = o & 63;
          size_t idx = (((size_t)(b * N_HEADC + h)) * SEQC + t) * D_KC + d;
          out0[idx] = f2bf(val + pbu[h * D_KC + d]);
          out1[idx] = f2bf(val + pbv[h * D_KC + d]);
        } else if (mode == MODE_KT) {
          int b = m >> 11, t = m & 2047;
          int h = o >> 6, d = o & 63;
          out0[(((size_t)(b * N_HEADC + h)) * D_KC + d) * SEQC + t] = f2bf(val);
        } else if (mode == MODE_V) {
          int b = m >> 11, t = m & 2047;
          int h = o >> 6, d = o & 63;
          out0[(((size_t)(b * N_HEADC + h)) * SEQC + t) * D_KC + d] = f2bf(val);
        } else if (mode == MODE_PT) {
          int h = o >> 6, d = o & 63;
          out0[((size_t)(h * D_KC + d)) * SEQC + m] = f2bf(val);
        } else {
          outf[(size_t)m * N_FEATC + o] = val;
        }
      }
    }
  }
}

// ---------------- fused flash-attention with TDM / async-LDS staging ----------------
// grid (T/128, B*H), 256 threads (8 waves). Wave owns 16 q rows.
// K/P tiles: TDM tensor_load_to_lds (wave 0). V tile: per-lane async loads.
// Double-buffered; 144B LDS row stride (TDM padding) for bank-friendly b128 reads.
__global__ __launch_bounds__(256) void attn_kernel(
    const u16* __restrict__ qu, const u16* __restrict__ qv,
    const u16* __restrict__ kt, const u16* __restrict__ pt,
    const u16* __restrict__ vb, u16* __restrict__ ao)
{
  __shared__ alignas(16) u16 kbuf[2][64 * 72];
  __shared__ alignas(16) u16 pbuf[2][64 * 72];
  __shared__ alignas(16) u16 vbuf[2][64 * 72];
  __shared__ alignas(16) u16 pstage[8 * 16 * 72];

  const int lane = threadIdx.x & 31;
  const int wave = threadIdx.x >> 5;
  const int tid  = threadIdx.x;
  const int bh = blockIdx.y;
  const int b = bh >> 3, h = bh & 7;
  const int t0 = blockIdx.x * 128 + wave * 16;
  const float LOG2E = 1.4426950408889634f;

  // Preload q_u / q_v A-fragments: 16 rows x 64 K -> 2 frags each
  v16bf au[2], av[2];
  {
    size_t rowb = ((size_t)bh * SEQC + t0 + (lane & 15)) * 128;
#pragma unroll
    for (int j = 0; j < 2; ++j) {
      au[j] = load_frag_a((const char*)qu + rowb + j * 64, lane);
      av[j] = load_frag_a((const char*)qv + rowb + j * 64, lane);
    }
  }

  float rowM[8], rowL[8];
  v8f O[4];
#pragma unroll
  for (int r = 0; r < 8; ++r) { rowM[r] = -1e30f; rowL[r] = 0.f; }
#pragma unroll
  for (int dt = 0; dt < 4; ++dt) O[dt] = (v8f){0.f, 0.f, 0.f, 0.f, 0.f, 0.f, 0.f, 0.f};

  u16* slab = &pstage[wave * 16 * 72];

  // ---- DMA issue helpers ----
  auto issue_kp = [&](int i) {   // wave 0 only: 2 TDM ops
    int par = i & 1;
    size_t s0 = (size_t)i * 64;
    tdm_load_2d((u32)(uintptr_t)&kbuf[par][0],
                (const char*)kt + (((size_t)bh * 64) * SEQC + s0) * 2, SEQC);
    tdm_load_2d((u32)(uintptr_t)&pbuf[par][0],
                (const char*)pt + (((size_t)h  * 64) * SEQC + s0) * 2, SEQC);
  };
  auto issue_v = [&](int i) {    // all threads: 2 async b128 each (8KB tile)
    int par = i & 1;
    size_t gbase = (u64)(uintptr_t)vb + (((size_t)bh * SEQC + (size_t)i * 64) * 128);
#pragma unroll
    for (int z = 0; z < 2; ++z) {
      int c = tid + z * 256;                       // 512 x 16B chunks
      u32 l = (u32)(uintptr_t)&vbuf[par][0] + (u32)((c >> 3) * 144 + (c & 7) * 16);
      u64 g = gbase + (size_t)c * 16;
      asm volatile("global_load_async_to_lds_b128 %0, %1, off"
                   :: "v"(l), "v"(g) : "memory");
    }
  };

  if (wave == 0) issue_kp(0);
  issue_v(0);

  for (int i = 0; i < SEQC / 64; ++i) {
    const int cur = i & 1;
    const bool more = (i + 1) < (SEQC / 64);
    if (more) {
      if (wave == 0) issue_kp(i + 1);
      issue_v(i + 1);
    }
    if (wave == 0) {
      if (more) __builtin_amdgcn_s_wait_tensorcnt(2);
      else      __builtin_amdgcn_s_wait_tensorcnt(0);
    }
    if (more) asm volatile("s_wait_asynccnt 0x2" ::: "memory");
    else      asm volatile("s_wait_asynccnt 0x0" ::: "memory");
    __syncthreads();

    const u16* kb = &kbuf[cur][0];
    const u16* pb = &pbuf[cur][0];
    const u16* vv = &vbuf[cur][0];

    // scores: S = (q_u k^T + q_v p^T) / 8   -> 4 col tiles of 16
    v8f sc[4];
#pragma unroll
    for (int c = 0; c < 4; ++c) sc[c] = (v8f){0.f, 0.f, 0.f, 0.f, 0.f, 0.f, 0.f, 0.f};
#pragma unroll
    for (int j = 0; j < 2; ++j) {
      const u16* krow = kb + (j * 32) * 72;
      const u16* prow = pb + (j * 32) * 72;
#pragma unroll
      for (int c = 0; c < 4; ++c) {
        v16bf kf = load_frag_b((const char*)(krow + c * 16), lane, 72);
        sc[c] = wmma_bf16(au[j], kf, sc[c]);
        v16bf pf = load_frag_b((const char*)(prow + c * 16), lane, 72);
        sc[c] = wmma_bf16(av[j], pf, sc[c]);
      }
    }
#pragma unroll
    for (int c = 0; c < 4; ++c)
#pragma unroll
      for (int r = 0; r < 8; ++r) sc[c][r] *= 0.125f;   // 1/sqrt(64)

    // online softmax (lanes 0-15 / 16-31 are the two row groups)
#pragma unroll
    for (int r = 0; r < 8; ++r) {
      float mx = fmaxf(fmaxf(sc[0][r], sc[1][r]), fmaxf(sc[2][r], sc[3][r]));
#pragma unroll
      for (int d = 1; d < 16; d <<= 1) mx = fmaxf(mx, __shfl_xor(mx, d, 32));
      float nM = fmaxf(rowM[r], mx);
      float rescale = exp2f((rowM[r] - nM) * LOG2E);
      rowM[r] = nM;
      float rs = 0.f;
#pragma unroll
      for (int c = 0; c < 4; ++c) {
        float p = exp2f((sc[c][r] - nM) * LOG2E);
        sc[c][r] = p;
        rs += p;
      }
#pragma unroll
      for (int d = 1; d < 16; d <<= 1) rs += __shfl_xor(rs, d, 32);
      rowL[r] = rowL[r] * rescale + rs;
#pragma unroll
      for (int dt = 0; dt < 4; ++dt) O[dt][r] *= rescale;
    }

    // stage P (C layout -> A layout) through wave-private LDS slab
#pragma unroll
    for (int c = 0; c < 4; ++c)
#pragma unroll
      for (int r = 0; r < 8; ++r)
        slab[(((lane >> 4) << 3) + r) * 72 + c * 16 + (lane & 15)] = f2bf(sc[c][r]);

    // O += P * V
#pragma unroll
    for (int j = 0; j < 2; ++j) {
      v16bf pfrag = load_frag_a((const char*)slab + (size_t)(lane & 15) * 144 + j * 64, lane);
      const u16* vrow = vv + (j * 32) * 72;
#pragma unroll
      for (int dt = 0; dt < 4; ++dt) {
        v16bf vf = load_frag_b((const char*)(vrow + dt * 16), lane, 72);
        O[dt] = wmma_bf16(pfrag, vf, O[dt]);
      }
    }
    __syncthreads();
  }

  // finalize: O / L, write bf16 to (b, t, h*d) for the output projection
#pragma unroll
  for (int r = 0; r < 8; ++r) {
    float inv = 1.f / rowL[r];
    int t = t0 + ((lane >> 4) << 3) + r;
#pragma unroll
    for (int dt = 0; dt < 4; ++dt) {
      float val = O[dt][r] * inv;
      ao[((size_t)(b * SEQC + t)) * N_FEATC + h * D_KC + dt * 16 + (lane & 15)] = f2bf(val);
    }
  }
}

// ---------------- host launcher ----------------
extern "C" void kernel_launch(void* const* d_in, const int* in_sizes, int n_in,
                              void* d_out, int out_size, void* d_ws, size_t ws_size,
                              hipStream_t stream) {
  const float* x    = (const float*)d_in[0];
  const float* pos  = (const float*)d_in[1];
  const float* Wq   = (const float*)d_in[2];
  const float* bq   = (const float*)d_in[3];
  const float* Wk   = (const float*)d_in[4];
  const float* bk   = (const float*)d_in[5];
  const float* Wv   = (const float*)d_in[6];
  const float* bv   = (const float*)d_in[7];
  const float* Wpos = (const float*)d_in[8];
  const float* Wout = (const float*)d_in[9];
  const float* bout = (const float*)d_in[10];
  const float* pbu  = (const float*)d_in[11];
  const float* pbv  = (const float*)d_in[12];
  float* outp = (float*)d_out;

  char* ws = (char*)d_ws;
  size_t off = 0;
  auto alloc = [&](size_t elems) -> u16* {
    u16* p = (u16*)(ws + off);
    off += elems * sizeof(u16);
    off = (off + 255) & ~(size_t)255;
    return p;
  };
  u16* xb  = alloc((size_t)BTC * N_FEATC);
  u16* pb  = alloc((size_t)SEQC * N_FEATC);
  u16* Wqt = alloc((size_t)N_FEATC * N_FEATC);
  u16* Wkt = alloc((size_t)N_FEATC * N_FEATC);
  u16* Wvt = alloc((size_t)N_FEATC * N_FEATC);
  u16* Wpt = alloc((size_t)N_FEATC * N_FEATC);
  u16* Wot = alloc((size_t)N_FEATC * N_FEATC);
  u16* quB = alloc((size_t)BTC * N_FEATC);
  u16* qvB = alloc((size_t)BTC * N_FEATC);
  u16* ktB = alloc((size_t)BTC * N_FEATC);
  u16* vbB = alloc((size_t)BTC * N_FEATC);
  u16* ptB = alloc((size_t)SEQC * N_FEATC);
  u16* aoB = alloc((size_t)BTC * N_FEATC);

  {
    int n = BTC * N_FEATC;
    cvt_bf16<<<dim3(2048), dim3(256), 0, stream>>>(x, xb, n);
    int np = SEQC * N_FEATC;
    cvt_bf16<<<dim3(1024), dim3(256), 0, stream>>>(pos, pb, np);
    int nw = N_FEATC * N_FEATC;
    dim3 gw((nw + 255) / 256);
    cvt_bf16_tr<<<gw, dim3(256), 0, stream>>>(Wq, Wqt);
    cvt_bf16_tr<<<gw, dim3(256), 0, stream>>>(Wk, Wkt);
    cvt_bf16_tr<<<gw, dim3(256), 0, stream>>>(Wv, Wvt);
    cvt_bf16_tr<<<gw, dim3(256), 0, stream>>>(Wpos, Wpt);
    cvt_bf16_tr<<<gw, dim3(256), 0, stream>>>(Wout, Wot);
  }

  dim3 blk(256);
  dim3 g1(BTC / 128, N_FEATC / 128);
  gemm_bf16_tn<<<g1, blk, 0, stream>>>(xb, Wqt, bq, MODE_QU, quB, qvB, pbu, pbv, nullptr);
  gemm_bf16_tn<<<g1, blk, 0, stream>>>(xb, Wkt, bk, MODE_KT, ktB, nullptr, nullptr, nullptr, nullptr);
  gemm_bf16_tn<<<g1, blk, 0, stream>>>(xb, Wvt, bv, MODE_V,  vbB, nullptr, nullptr, nullptr, nullptr);
  dim3 g2(SEQC / 128, N_FEATC / 128);
  gemm_bf16_tn<<<g2, blk, 0, stream>>>(pb, Wpt, nullptr, MODE_PT, ptB, nullptr, nullptr, nullptr, nullptr);

  attn_kernel<<<dim3(SEQC / 128, BATCHC * N_HEADC), blk, 0, stream>>>(quB, qvB, ktB, ptB, vbB, aoB);

  gemm_bf16_tn<<<g1, blk, 0, stream>>>(aoB, Wot, bout, MODE_OUT, nullptr, nullptr, nullptr, nullptr, outp);
}
